// BRNNModel_19816979104407
// MI455X (gfx1250) — compile-verified, matrix operand order checked by reading, not run on hardware
//
#include <hip/hip_runtime.h>
#include <hip/hip_bf16.h>

typedef __attribute__((ext_vector_type(16))) __bf16 v16bf;
typedef __attribute__((ext_vector_type(8)))  __bf16 v8bf;
typedef __attribute__((ext_vector_type(8)))  float  v8f;

#define VOCAB  10000
#define EMB    128
#define UNITS  256
#define BATCH  2048
#define TLEN   256
#define KTOT   384          // EMB + UNITS
#define NGATES 1024         // 4*UNITS
#define MTILE  32           // batch rows per workgroup
#define XPAD   136          // 128 + 8 halves (bank-conflict pad)
#define HPAD   264          // 256 + 8 halves

static __device__ __forceinline__ float sigm(float x) {
    return 1.0f / (1.0f + __expf(-x));
}
static __device__ __forceinline__ float tanh_fast(float x) {
    // tanh(x) = 2*sigmoid(2x) - 1
    return 2.0f / (1.0f + __expf(-2.0f * x)) - 1.0f;
}

// ---------------- conversion / packing kernels ----------------

__global__ void cvt_bf16_kernel(const float* __restrict__ in,
                                __bf16* __restrict__ out, int n) {
    int i = blockIdx.x * 256 + threadIdx.x;
    if (i < n) out[i] = (__bf16)in[i];
}

// Build packed transposed weights: out[n*KTOT + k] = (k<EMB ? W[k][n] : U[k-EMB][n]) as bf16
__global__ void pack_wu_kernel(const float* __restrict__ W,
                               const float* __restrict__ U,
                               __bf16* __restrict__ out) {
    int i = blockIdx.x * 256 + threadIdx.x;
    if (i < NGATES * KTOT) {
        int n = i / KTOT, k = i % KTOT;
        float v = (k < EMB) ? W[(size_t)k * NGATES + n]
                            : U[(size_t)(k - EMB) * NGATES + n];
        out[i] = (__bf16)v;
    }
}

// ---------------- fused bidirectional LSTM kernel ----------------
// grid = (BATCH/MTILE, 2), block = 256 threads (8 waves).
// Each block: one 32-row batch tile, one direction, all T steps.

__global__ __launch_bounds__(256)
void lstm_kernel(const int* __restrict__ tokens,
                 const __bf16* __restrict__ embb,
                 const __bf16* __restrict__ wu_f,
                 const __bf16* __restrict__ wu_b,
                 const float* __restrict__ bias_f,
                 const float* __restrict__ bias_b,
                 float* __restrict__ hout_f,
                 float* __restrict__ hout_b)
{
    __shared__ int    tok[MTILE * TLEN];       // 32 KB
    __shared__ __bf16 xs[MTILE * XPAD];        // ~8.5 KB
    __shared__ __bf16 hs[MTILE * HPAD];        // ~16.5 KB

    const int tid  = threadIdx.x;
    const int wave = tid >> 5;
    const int lane = tid & 31;
    const int l15  = lane & 15;
    const int hi   = lane >> 4;            // which K-half / which M-half
    const int dir  = blockIdx.y;
    const int r0   = blockIdx.x * MTILE;

    const __bf16* wu   = dir ? wu_b   : wu_f;
    const float*  bia  = dir ? bias_b : bias_f;
    float*        hout = (dir ? hout_b : hout_f) + (size_t)r0 * UNITS;

    // stage token tile, zero h
    for (int i = tid; i < MTILE * TLEN; i += 256)
        tok[i] = tokens[(size_t)r0 * TLEN + i];
    for (int i = tid; i < MTILE * HPAD; i += 256)
        hs[i] = (__bf16)0.0f;

    const int wcol = wave * 32;            // this wave's 32 gate-local columns

    // B-matrix row pointers + bias per (gate, colpair) tile; col fixed per lane
    const __bf16* bptr[4][2];
    float bval[4][2];
#pragma unroll
    for (int g = 0; g < 4; ++g)
#pragma unroll
        for (int cp = 0; cp < 2; ++cp) {
            int n = g * UNITS + wcol + cp * 16 + l15;
            bptr[g][cp] = wu + (size_t)n * KTOT;
            bval[g][cp] = bia[n];
        }

    // persistent cell / hidden state in registers
    float creg[2][2][8];
    float hreg[2][2][8];
#pragma unroll
    for (int m = 0; m < 2; ++m)
#pragma unroll
        for (int cp = 0; cp < 2; ++cp)
#pragma unroll
            for (int r = 0; r < 8; ++r) { creg[m][cp][r] = 0.f; hreg[m][cp][r] = 0.f; }

    __syncthreads();

    for (int t = 0; t < TLEN; ++t) {
        const int ts = dir ? (TLEN - 1 - t) : t;

        // gather x_t tile: 32 rows x 128 bf16; each thread copies 16 halves
        {
            int row = tid >> 3, seg = tid & 7;
            int tv  = tok[row * TLEN + ts];
            const v8bf* src = (const v8bf*)(embb + (size_t)tv * EMB + seg * 16);
            v8bf* dst = (v8bf*)(xs + row * XPAD + seg * 16);
            dst[0] = src[0];
            dst[1] = src[1];
        }
        __syncthreads();

        // accumulators seeded with per-column bias
        v8f acc[2][4][2];
#pragma unroll
        for (int m = 0; m < 2; ++m)
#pragma unroll
            for (int g = 0; g < 4; ++g)
#pragma unroll
                for (int cp = 0; cp < 2; ++cp) {
                    v8f z;
#pragma unroll
                    for (int e = 0; e < 8; ++e) z[e] = bval[g][cp];
                    acc[m][g][cp] = z;
                }

        // K loop: z = [x_t | h] @ [W;U]^T  (K = 384 in 12 chunks of 32)
#pragma unroll 2
        for (int kc = 0; kc < 12; ++kc) {
            const int k = kc * 32;

            v16bf afrag[2];
#pragma unroll
            for (int m = 0; m < 2; ++m) {
                int row = m * 16 + l15;
                const __bf16* ap = (k < EMB) ? (xs + row * XPAD + k)
                                             : (hs + row * HPAD + (k - EMB));
                v8bf lo = *(const v8bf*)(ap + hi * 8);
                v8bf hi8 = *(const v8bf*)(ap + 16 + hi * 8);
                v16bf a;
#pragma unroll
                for (int e = 0; e < 8; ++e) { a[e] = lo[e]; a[8 + e] = hi8[e]; }
                afrag[m] = a;
            }

#pragma unroll
            for (int g = 0; g < 4; ++g)
#pragma unroll
                for (int cp = 0; cp < 2; ++cp) {
                    const __bf16* bp = bptr[g][cp] + k;
                    v8bf lo = *(const v8bf*)(bp + hi * 8);
                    v8bf hi8 = *(const v8bf*)(bp + 16 + hi * 8);
                    v16bf b;
#pragma unroll
                    for (int e = 0; e < 8; ++e) { b[e] = lo[e]; b[8 + e] = hi8[e]; }
                    acc[0][g][cp] = __builtin_amdgcn_wmma_f32_16x16x32_bf16(
                        false, afrag[0], false, b, (short)0, acc[0][g][cp], false, false);
                    acc[1][g][cp] = __builtin_amdgcn_wmma_f32_16x16x32_bf16(
                        false, afrag[1], false, b, (short)0, acc[1][g][cp], false, false);
                }
        }

        __syncthreads();   // all hs reads finished before overwrite

        // gates + masked state update (C/D layout: row = m*16 + hi*8 + r, col = lane&15)
#pragma unroll
        for (int m = 0; m < 2; ++m)
#pragma unroll
            for (int r = 0; r < 8; ++r) {
                int row = m * 16 + hi * 8 + r;
                bool msk = tok[row * TLEN + ts] != 0;
#pragma unroll
                for (int cp = 0; cp < 2; ++cp) {
                    float iv = sigm(acc[m][0][cp][r]);
                    float fv = sigm(acc[m][1][cp][r]);
                    float gv = tanh_fast(acc[m][2][cp][r]);
                    float ov = sigm(acc[m][3][cp][r]);
                    float cn = fv * creg[m][cp][r] + iv * gv;
                    float hn = ov * tanh_fast(cn);
                    if (msk) { creg[m][cp][r] = cn; hreg[m][cp][r] = hn; }
                }
            }

        // write h back to LDS as bf16 for next step's A fragments
#pragma unroll
        for (int m = 0; m < 2; ++m)
#pragma unroll
            for (int cp = 0; cp < 2; ++cp) {
                int col = wcol + cp * 16 + l15;
#pragma unroll
                for (int r = 0; r < 8; ++r) {
                    int row = m * 16 + hi * 8 + r;
                    hs[row * HPAD + col] = (__bf16)hreg[m][cp][r];
                }
            }
        __syncthreads();
    }

    // final hidden state -> global (fp32)
#pragma unroll
    for (int m = 0; m < 2; ++m)
#pragma unroll
        for (int cp = 0; cp < 2; ++cp) {
            int col = wcol + cp * 16 + l15;
#pragma unroll
            for (int r = 0; r < 8; ++r) {
                int row = m * 16 + hi * 8 + r;
                hout[(size_t)row * UNITS + col] = hreg[m][cp][r];
            }
        }
}

// ---------------- head: relu(h@W1+b1)@W2+b2 -> sigmoid ----------------
// one wave per batch row; lane owns 4 of 128 hidden units

__global__ __launch_bounds__(256)
void head_kernel(const float* __restrict__ hf, const float* __restrict__ hb,
                 const float* __restrict__ W1, const float* __restrict__ b1,
                 const float* __restrict__ W2, const float* __restrict__ b2,
                 float* __restrict__ out)
{
    int row  = blockIdx.x * 8 + (threadIdx.x >> 5);
    int lane = threadIdx.x & 31;
    if (row >= BATCH) return;

    float acc[4] = {0.f, 0.f, 0.f, 0.f};
    const float* h0 = hf + (size_t)row * UNITS;
    const float* h1 = hb + (size_t)row * UNITS;
    for (int k = 0; k < 2 * UNITS; ++k) {
        float hv = (k < UNITS) ? h0[k] : h1[k - UNITS];
        const float* wrow = W1 + (size_t)k * 128;
#pragma unroll
        for (int j = 0; j < 4; ++j)
            acc[j] = fmaf(hv, wrow[lane + j * 32], acc[j]);
    }
    float s = 0.f;
#pragma unroll
    for (int j = 0; j < 4; ++j) {
        int col = lane + j * 32;
        float d = fmaxf(acc[j] + b1[col], 0.f);
        s = fmaf(d, W2[col], s);
    }
#pragma unroll
    for (int off = 16; off > 0; off >>= 1)
        s += __shfl_down(s, off, 32);
    if (lane == 0)
        out[row] = 1.0f / (1.0f + __expf(-(s + b2[0])));
}

// ---------------- launcher ----------------

extern "C" void kernel_launch(void* const* d_in, const int* in_sizes, int n_in,
                              void* d_out, int out_size, void* d_ws, size_t ws_size,
                              hipStream_t stream) {
    const int*   tokens = (const int*)  d_in[0];
    const float* emb    = (const float*)d_in[1];
    const float* Wf     = (const float*)d_in[2];
    const float* Uf     = (const float*)d_in[3];
    const float* bf     = (const float*)d_in[4];
    const float* Wb     = (const float*)d_in[5];
    const float* Ub     = (const float*)d_in[6];
    const float* bb     = (const float*)d_in[7];
    const float* W1     = (const float*)d_in[8];
    const float* b1     = (const float*)d_in[9];
    const float* W2     = (const float*)d_in[10];
    const float* b2     = (const float*)d_in[11];
    float* out = (float*)d_out;

    char* ws = (char*)d_ws;
    size_t o = 0;
    __bf16* embb = (__bf16*)(ws + o); o += (size_t)VOCAB * EMB * sizeof(__bf16);
    o = (o + 255) & ~(size_t)255;
    __bf16* wu_f = (__bf16*)(ws + o); o += (size_t)NGATES * KTOT * sizeof(__bf16);
    __bf16* wu_b = (__bf16*)(ws + o); o += (size_t)NGATES * KTOT * sizeof(__bf16);
    float*  h_f  = (float*) (ws + o); o += (size_t)BATCH * UNITS * sizeof(float);
    float*  h_b  = (float*) (ws + o); o += (size_t)BATCH * UNITS * sizeof(float);

    {
        int n = VOCAB * EMB;
        cvt_bf16_kernel<<<(n + 255) / 256, 256, 0, stream>>>(emb, embb, n);
    }
    {
        int n = NGATES * KTOT;
        pack_wu_kernel<<<(n + 255) / 256, 256, 0, stream>>>(Wf, Uf, wu_f);
        pack_wu_kernel<<<(n + 255) / 256, 256, 0, stream>>>(Wb, Ub, wu_b);
    }
    {
        dim3 grid(BATCH / MTILE, 2);
        lstm_kernel<<<grid, 256, 0, stream>>>(tokens, embb, wu_f, wu_b,
                                              bf, bb, h_f, h_b);
    }
    {
        head_kernel<<<BATCH / 8, 256, 0, stream>>>(h_f, h_b, W1, b1, W2, b2, out);
    }
}
